// Transformer_55748675502916
// MI455X (gfx1250) — compile-verified, matrix operand order checked by reading, not run on hardware
//
#include <hip/hip_runtime.h>
#include <hip/hip_bf16.h>
#include <math.h>

// ---------------------------------------------------------------------------
// Swin transformer block on gfx1250 (MI455X)
//   - all GEMMs via v_wmma_f32_16x16x32_f16, 32x64 register-blocked per wave
//   - attention fused in LDS; K / V^T slabs staged via global_load_async_to_lds
// B=64, N=256 tokens, C=256, heads=8, head_dim=32, hidden=1024
// ---------------------------------------------------------------------------

typedef __attribute__((ext_vector_type(16))) _Float16 v16h;
typedef __attribute__((ext_vector_type(8)))  _Float16 v8h;
typedef __attribute__((ext_vector_type(8)))  float    v8f;

#define BATCH   64
#define NTOK    256
#define DIM     256
#define HEADS   8
#define HDIM    32
#define HIDDEN  1024
#define ROWS_TOT (BATCH * NTOK)   // 16384

// ---- wave32 reductions ----
__device__ __forceinline__ float warp_sum(float v) {
#pragma unroll
    for (int m = 16; m > 0; m >>= 1) v += __shfl_xor(v, m, 32);
    return v;
}
__device__ __forceinline__ float warp_max(float v) {
#pragma unroll
    for (int m = 16; m > 0; m >>= 1) v = fmaxf(v, __shfl_xor(v, m, 32));
    return v;
}

// ---- CDNA5 async global->LDS copy (ASYNCcnt-tracked), 16B per lane ----
__device__ __forceinline__ void async_g2l_b128(void* lds_ptr, const void* gptr) {
    unsigned ldsoff = (unsigned)(size_t)lds_ptr;   // flat LDS addr low 32 bits = LDS offset
    asm volatile("global_load_async_to_lds_b128 %0, %1, off"
                 :: "v"(ldsoff), "v"(gptr) : "memory");
}
__device__ __forceinline__ void wait_async0() {
    asm volatile("s_wait_asynccnt 0x0" ::: "memory");
}

// ---- WMMA fragment load (16-bit A-layout, ISA 7.12.2) from row-major f16 ----
// lane L holds row (L&15); K elements: (L>>4)*8 + 0..7  and  16 + (L>>4)*8 + 0..7
__device__ __forceinline__ v16h load_frag(const _Float16* p, int ld) {
    int lane = threadIdx.x & 31;
    const _Float16* q = p + (lane & 15) * ld + ((lane >> 4) * 8);
    v8h lo = *(const v8h*)(q);
    v8h hi = *(const v8h*)(q + 16);
    v16h r;
#pragma unroll
    for (int i = 0; i < 8; ++i) { r[i] = lo[i]; r[i + 8] = hi[i]; }
    return r;
}

__device__ __forceinline__ v8f wmma_one(v16h a, v16h b, v8f c) {
    return __builtin_amdgcn_wmma_f32_16x16x32_f16(false, a, false, b,
                                                  (short)0, c, false, false);
}

// ---- 32x64 output block per wave: 2 M-tiles x 4 N-tiles, 6 loads : 8 wmma ----
template <int KTOT>
__device__ __forceinline__ void wmma_block_2x4(const _Float16* A, int lda,
                                               const _Float16* B, int ldb,
                                               v8f acc[2][4]) {
    for (int k = 0; k < KTOT; k += 32) {
        v16h a0 = load_frag(A + k, lda);
        v16h a1 = load_frag(A + 16 * lda + k, lda);
        v16h b0 = load_frag(B + k, ldb);
        v16h b1 = load_frag(B + 16 * ldb + k, ldb);
        v16h b2 = load_frag(B + 32 * ldb + k, ldb);
        v16h b3 = load_frag(B + 48 * ldb + k, ldb);
        acc[0][0] = wmma_one(a0, b0, acc[0][0]);
        acc[0][1] = wmma_one(a0, b1, acc[0][1]);
        acc[0][2] = wmma_one(a0, b2, acc[0][2]);
        acc[0][3] = wmma_one(a0, b3, acc[0][3]);
        acc[1][0] = wmma_one(a1, b0, acc[1][0]);
        acc[1][1] = wmma_one(a1, b1, acc[1][1]);
        acc[1][2] = wmma_one(a1, b2, acc[1][2]);
        acc[1][3] = wmma_one(a1, b3, acc[1][3]);
    }
}

// ---------------------------------------------------------------------------
// f32 -> f16 weight conversion
// ---------------------------------------------------------------------------
__global__ void swin_cvt_f16(const float* __restrict__ w, _Float16* __restrict__ o, int n) {
    int i = blockIdx.x * blockDim.x + threadIdx.x;
    if (i < n) o[i] = (_Float16)w[i];
}

// ---------------------------------------------------------------------------
// (B,C,H,W) -> (B,N,C) f32 shortcut + LayerNorm1 -> f16. Wave per row.
// ---------------------------------------------------------------------------
__global__ void swin_ln1(const float* __restrict__ x, const float* __restrict__ g,
                         const float* __restrict__ bt, float* __restrict__ xf,
                         _Float16* __restrict__ xn) {
    int row  = blockIdx.x * 8 + (threadIdx.x >> 5);
    int lane = threadIdx.x & 31;
    int b = row >> 8, tok = row & 255;
    const float* xb = x + (size_t)b * DIM * NTOK;
    float v[8]; float s = 0.f;
#pragma unroll
    for (int j = 0; j < 8; ++j) {
        int c = j * 32 + lane;
        v[j] = xb[c * NTOK + tok];
        s += v[j];
    }
    float mu = warp_sum(s) * (1.f / 256.f);
    float s2 = 0.f;
#pragma unroll
    for (int j = 0; j < 8; ++j) { float d = v[j] - mu; s2 += d * d; }
    float rstd = rsqrtf(warp_sum(s2) * (1.f / 256.f) + 1e-5f);
    float* xfr = xf + (size_t)row * DIM;
    _Float16* xnr = xn + (size_t)row * DIM;
#pragma unroll
    for (int j = 0; j < 8; ++j) {
        int c = j * 32 + lane;
        xfr[c] = v[j];
        xnr[c] = (_Float16)((v[j] - mu) * rstd * g[c] + bt[c]);
    }
}

// ---------------------------------------------------------------------------
// LayerNorm2 (contiguous f32 in -> f16 out). Wave per row.
// ---------------------------------------------------------------------------
__global__ void swin_ln2(const float* __restrict__ x1, const float* __restrict__ g,
                         const float* __restrict__ bt, _Float16* __restrict__ xn) {
    int row  = blockIdx.x * 8 + (threadIdx.x >> 5);
    int lane = threadIdx.x & 31;
    const float* xr = x1 + (size_t)row * DIM;
    float v[8]; float s = 0.f;
#pragma unroll
    for (int j = 0; j < 8; ++j) { v[j] = xr[j * 32 + lane]; s += v[j]; }
    float mu = warp_sum(s) * (1.f / 256.f);
    float s2 = 0.f;
#pragma unroll
    for (int j = 0; j < 8; ++j) { float d = v[j] - mu; s2 += d * d; }
    float rstd = rsqrtf(warp_sum(s2) * (1.f / 256.f) + 1e-5f);
    _Float16* xnr = xn + (size_t)row * DIM;
#pragma unroll
    for (int j = 0; j < 8; ++j) {
        int c = j * 32 + lane;
        xnr[c] = (_Float16)((v[j] - mu) * rstd * g[c] + bt[c]);
    }
}

// ---------------------------------------------------------------------------
// QKV: xn(16384x256) @ qkv_w^T(256x768) + b  -> q (scaled), k, v^T  (all f16)
// wave block 32x64; grid: 64 * 8(Mb) * 12(Nb) / 8 waves = 768 blocks
// ---------------------------------------------------------------------------
__global__ void swin_qkv(const _Float16* __restrict__ xn, const _Float16* __restrict__ wq,
                         const float* __restrict__ qb, _Float16* __restrict__ qh,
                         _Float16* __restrict__ kh, _Float16* __restrict__ vt) {
    int wb = blockIdx.x * 8 + (threadIdx.x >> 5);     // [0, 6144)
    int b   = wb / 96;
    int rem = wb % 96;
    int tm2 = rem / 12, tn2 = rem % 12;
    const _Float16* A = xn + ((size_t)b * NTOK + tm2 * 32) * DIM;
    const _Float16* B = wq + (size_t)tn2 * 64 * DIM;
    v8f acc[2][4] = {};
    wmma_block_2x4<DIM>(A, DIM, B, DIM, acc);

    int lane = threadIdx.x & 31;
    int nl = lane & 15;
#pragma unroll
    for (int j = 0; j < 4; ++j) {
        int n = tn2 * 64 + j * 16 + nl;               // [0,768)
        int which = n >> 8, cc = n & 255;
        int h = cc >> 5, dd = cc & 31;
        float bias = qb[n];
        size_t bh = (size_t)(b * HEADS + h);
#pragma unroll
        for (int i = 0; i < 2; ++i) {
#pragma unroll
            for (int r = 0; r < 8; ++r) {
                int m = tm2 * 32 + i * 16 + (lane >> 4) * 8 + r;   // token
                float val = acc[i][j][r] + bias;
                if (which == 0)      qh[(bh * NTOK + m) * HDIM + dd] = (_Float16)(val * 0.1767766953f);
                else if (which == 1) kh[(bh * NTOK + m) * HDIM + dd] = (_Float16)val;
                else                 vt[(bh * HDIM + dd) * NTOK + m] = (_Float16)val;
            }
        }
    }
}

// ---------------------------------------------------------------------------
// Fused attention per (b,h,64-row slab):
//   - K-slab (256x32) and V^T-slab (32x256) staged to LDS via async b128 copies
//   - S = q@k^T + rel_bias into LDS f32 (WMMA)
//   - softmax rows -> f16 P in LDS
//   - out = P@v (WMMA, both operands from LDS)
// grid = B*H*4 blocks of 256 threads (8 waves). LDS = 128 KB/block.
// ---------------------------------------------------------------------------
__global__ void swin_attn(const _Float16* __restrict__ qh, const _Float16* __restrict__ kh,
                          const _Float16* __restrict__ vt, const float* __restrict__ btab,
                          _Float16* __restrict__ ao) {
    __shared__ __align__(16) float    S [64 * 256];   // 64 KB scores
    __shared__ __align__(16) _Float16 P [64 * 256];   // 32 KB probs (f16)
    __shared__ __align__(16) _Float16 Kl[256 * HDIM]; // 16 KB K slab
    __shared__ __align__(16) _Float16 Vl[HDIM * 256]; // 16 KB V^T slab

    int part = blockIdx.x & 3;
    int bh   = blockIdx.x >> 2;
    int b = bh >> 3, h = bh & 7;
    int r0 = part * 64;
    int w    = threadIdx.x >> 5;
    int lane = threadIdx.x & 31;
    int nl = lane & 15;

    const _Float16* Q  = qh + (size_t)bh * NTOK * HDIM;
    const _Float16* Kg = kh + (size_t)bh * NTOK * HDIM;
    const _Float16* Vg = vt + (size_t)bh * HDIM * NTOK;

    // ---- stage K and V^T slabs into LDS: 2 x 16KB, 16B per thread per step
    {
        int t = threadIdx.x;
#pragma unroll
        for (int c = 0; c < 4; ++c) {
            int chunk = (t + c * 256) * 8;            // in halves
            async_g2l_b128(&Kl[chunk], Kg + chunk);
            async_g2l_b128(&Vl[chunk], Vg + chunk);
        }
        wait_async0();
    }
    __syncthreads();

    // ---- scores: 4 strips of 16 rows; waves (2s,2s+1) split the 16 col-tiles
    {
        int s = w >> 1, jhalf = w & 1;
        v16h a = load_frag(Q + (size_t)(r0 + s * 16) * HDIM, HDIM);
#pragma unroll
        for (int jj = 0; jj < 8; ++jj) {
            int jt = jhalf * 8 + jj;
            v16h bf = load_frag(Kl + jt * 16 * HDIM, HDIM);
            v8f acc = {};
            acc = wmma_one(a, bf, acc);
            int j = jt * 16 + nl;
            int jy = j >> 4, jx = j & 15;
#pragma unroll
            for (int r = 0; r < 8; ++r) {
                int lr = s * 16 + (lane >> 4) * 8 + r;
                int i  = r0 + lr;
                int iy = i >> 4, ix = i & 15;
                int idx = (iy - jy + 15) * 31 + (ix - jx + 15);
                S[lr * 256 + j] = acc[r] + btab[idx * HEADS + h];
            }
        }
    }
    __syncthreads();

    // ---- softmax: 8 rows per wave, 8 cols per lane; write f16 P
#pragma unroll
    for (int rr = 0; rr < 8; ++rr) {
        int lr = w * 8 + rr;
        const float* row = S + lr * 256;
        _Float16*   prow = P + lr * 256;
        float vv[8]; float mx = -3.0e38f;
#pragma unroll
        for (int j = 0; j < 8; ++j) { vv[j] = row[j * 32 + lane]; mx = fmaxf(mx, vv[j]); }
        mx = warp_max(mx);
        float sum = 0.f;
#pragma unroll
        for (int j = 0; j < 8; ++j) { vv[j] = expf(vv[j] - mx); sum += vv[j]; }
        float inv = 1.f / warp_sum(sum);
#pragma unroll
        for (int j = 0; j < 8; ++j) prow[j * 32 + lane] = (_Float16)(vv[j] * inv);
    }
    __syncthreads();

    // ---- out = P @ v : 4 row-tiles x 2 col-tiles; wave w -> (w>>1, w&1)
    {
        int rt = w >> 1, ct = w & 1;
        v8f acc = {};
        for (int k = 0; k < 256; k += 32) {
            v16h a  = load_frag(P + (rt * 16) * 256 + k, 256);
            v16h bf = load_frag(Vl + ct * 16 * 256 + k, 256);
            acc = wmma_one(a, bf, acc);
        }
#pragma unroll
        for (int r = 0; r < 8; ++r) {
            int tok = r0 + rt * 16 + (lane >> 4) * 8 + r;
            int c   = h * HDIM + ct * 16 + nl;
            ao[((size_t)b * NTOK + tok) * DIM + c] = (_Float16)acc[r];
        }
    }
}

// ---------------------------------------------------------------------------
// proj GEMM + bias + residual -> x1 (f32).  wave block 32x64; 256 blocks
// ---------------------------------------------------------------------------
__global__ void swin_proj(const _Float16* __restrict__ ao, const _Float16* __restrict__ wp,
                          const float* __restrict__ pb, const float* __restrict__ sc,
                          float* __restrict__ x1) {
    int wb = blockIdx.x * 8 + (threadIdx.x >> 5);     // [0, 2048)
    int tm2 = wb >> 2, tn2 = wb & 3;
    v8f acc[2][4] = {};
    wmma_block_2x4<DIM>(ao + (size_t)tm2 * 32 * DIM, DIM,
                        wp + (size_t)tn2 * 64 * DIM, DIM, acc);
    int lane = threadIdx.x & 31;
    int nl = lane & 15;
#pragma unroll
    for (int j = 0; j < 4; ++j) {
        int n = tn2 * 64 + j * 16 + nl;
        float bias = pb[n];
#pragma unroll
        for (int i = 0; i < 2; ++i) {
#pragma unroll
            for (int r = 0; r < 8; ++r) {
                int m = tm2 * 32 + i * 16 + (lane >> 4) * 8 + r;
                size_t idx = (size_t)m * DIM + n;
                x1[idx] = acc[i][j][r] + bias + sc[idx];
            }
        }
    }
}

// ---------------------------------------------------------------------------
// fc1 GEMM + bias + exact GELU -> f16 (16384 x 1024).  1024 blocks
// ---------------------------------------------------------------------------
__global__ void swin_fc1(const _Float16* __restrict__ xn, const _Float16* __restrict__ w1,
                         const float* __restrict__ b1, _Float16* __restrict__ gh) {
    int wb = blockIdx.x * 8 + (threadIdx.x >> 5);     // [0, 8192)
    int tm2 = wb >> 4, tn2 = wb & 15;
    v8f acc[2][4] = {};
    wmma_block_2x4<DIM>(xn + (size_t)tm2 * 32 * DIM, DIM,
                        w1 + (size_t)tn2 * 64 * DIM, DIM, acc);
    int lane = threadIdx.x & 31;
    int nl = lane & 15;
#pragma unroll
    for (int j = 0; j < 4; ++j) {
        int n = tn2 * 64 + j * 16 + nl;
        float bias = b1[n];
#pragma unroll
        for (int i = 0; i < 2; ++i) {
#pragma unroll
            for (int r = 0; r < 8; ++r) {
                int m = tm2 * 32 + i * 16 + (lane >> 4) * 8 + r;
                float v = acc[i][j][r] + bias;
                v = 0.5f * v * (1.f + erff(v * 0.70710678118f));
                gh[(size_t)m * HIDDEN + n] = (_Float16)v;
            }
        }
    }
}

// ---------------------------------------------------------------------------
// fc2 GEMM (K=1024) + bias + residual, write NCHW f32 output.  256 blocks
// ---------------------------------------------------------------------------
__global__ void swin_fc2(const _Float16* __restrict__ gh, const _Float16* __restrict__ w2,
                         const float* __restrict__ b2, const float* __restrict__ x1,
                         float* __restrict__ out) {
    int wb = blockIdx.x * 8 + (threadIdx.x >> 5);     // [0, 2048)
    int tm2 = wb >> 2, tn2 = wb & 3;
    v8f acc[2][4] = {};
    wmma_block_2x4<HIDDEN>(gh + (size_t)tm2 * 32 * HIDDEN, HIDDEN,
                           w2 + (size_t)tn2 * 64 * HIDDEN, HIDDEN, acc);
    int lane = threadIdx.x & 31;
    int nl = lane & 15;
#pragma unroll
    for (int j = 0; j < 4; ++j) {
        int n = tn2 * 64 + j * 16 + nl;
        float bias = b2[n];
#pragma unroll
        for (int i = 0; i < 2; ++i) {
#pragma unroll
            for (int r = 0; r < 8; ++r) {
                int m = tm2 * 32 + i * 16 + (lane >> 4) * 8 + r;  // b*256+tok
                float v = acc[i][j][r] + bias + x1[(size_t)m * DIM + n];
                int b = m >> 8, tok = m & 255;
                out[((size_t)b * DIM + n) * NTOK + tok] = v;      // (B,C,H*W)
            }
        }
    }
}

// ---------------------------------------------------------------------------
extern "C" void kernel_launch(void* const* d_in, const int* in_sizes, int n_in,
                              void* d_out, int out_size, void* d_ws, size_t ws_size,
                              hipStream_t stream) {
    const float* x        = (const float*)d_in[0];
    const float* n1g      = (const float*)d_in[1];
    const float* n1b      = (const float*)d_in[2];
    const float* qkv_w    = (const float*)d_in[3];
    const float* qkv_b    = (const float*)d_in[4];
    const float* rel_tab  = (const float*)d_in[5];
    const float* proj_w   = (const float*)d_in[6];
    const float* proj_b   = (const float*)d_in[7];
    const float* n2g      = (const float*)d_in[8];
    const float* n2b      = (const float*)d_in[9];
    const float* fc1_w    = (const float*)d_in[10];
    const float* fc1_b    = (const float*)d_in[11];
    const float* fc2_w    = (const float*)d_in[12];
    const float* fc2_b    = (const float*)d_in[13];
    float* out = (float*)d_out;

    char* p = (char*)d_ws;
    auto alloc = [&](size_t bytes) {
        void* r = (void*)p;
        p += (bytes + 255) & ~(size_t)255;
        return r;
    };
    _Float16* wq_h = (_Float16*)alloc((size_t)3 * DIM * DIM * 2);
    _Float16* wp_h = (_Float16*)alloc((size_t)DIM * DIM * 2);
    _Float16* w1_h = (_Float16*)alloc((size_t)HIDDEN * DIM * 2);
    _Float16* w2_h = (_Float16*)alloc((size_t)DIM * HIDDEN * 2);
    float*    xf   = (float*)   alloc((size_t)ROWS_TOT * DIM * 4);
    _Float16* xn   = (_Float16*)alloc((size_t)ROWS_TOT * DIM * 2);
    _Float16* qh   = (_Float16*)alloc((size_t)BATCH * HEADS * NTOK * HDIM * 2);
    _Float16* kh   = (_Float16*)alloc((size_t)BATCH * HEADS * NTOK * HDIM * 2);
    _Float16* vt   = (_Float16*)alloc((size_t)BATCH * HEADS * HDIM * NTOK * 2);
    _Float16* ao   = (_Float16*)alloc((size_t)ROWS_TOT * DIM * 2);
    float*    x1   = (float*)   alloc((size_t)ROWS_TOT * DIM * 4);
    _Float16* x1n  = (_Float16*)alloc((size_t)ROWS_TOT * DIM * 2);
    _Float16* gh   = (_Float16*)alloc((size_t)ROWS_TOT * HIDDEN * 2);
    (void)ws_size; (void)n_in; (void)in_sizes; (void)out_size;

    swin_cvt_f16<<<(3 * DIM * DIM + 255) / 256, 256, 0, stream>>>(qkv_w, wq_h, 3 * DIM * DIM);
    swin_cvt_f16<<<(DIM * DIM + 255) / 256, 256, 0, stream>>>(proj_w, wp_h, DIM * DIM);
    swin_cvt_f16<<<(HIDDEN * DIM + 255) / 256, 256, 0, stream>>>(fc1_w, w1_h, HIDDEN * DIM);
    swin_cvt_f16<<<(DIM * HIDDEN + 255) / 256, 256, 0, stream>>>(fc2_w, w2_h, DIM * HIDDEN);

    swin_ln1<<<ROWS_TOT / 8, 256, 0, stream>>>(x, n1g, n1b, xf, xn);
    swin_qkv<<<BATCH * 8 * 12 / 8, 256, 0, stream>>>(xn, wq_h, qkv_b, qh, kh, vt);
    swin_attn<<<BATCH * HEADS * 4, 256, 0, stream>>>(qh, kh, vt, rel_tab, ao);
    swin_proj<<<(ROWS_TOT / 32) * 4 / 8, 256, 0, stream>>>(ao, wp_h, proj_b, xf, x1);
    swin_ln2<<<ROWS_TOT / 8, 256, 0, stream>>>(x1, n2g, n2b, x1n);
    swin_fc1<<<(ROWS_TOT / 32) * 16 / 8, 256, 0, stream>>>(x1n, w1_h, fc1_b, gh);
    swin_fc2<<<(ROWS_TOT / 32) * 4 / 8, 256, 0, stream>>>(gh, w2_h, fc2_b, x1, out);
}
